// BitGQAAttention_36378372997522
// MI455X (gfx1250) — 16-bit and 8-bit matrix operand order repaired and checked by reading, compiled, not run on hardware
//
#include <hip/hip_runtime.h>
#include <hip/hip_bf16.h>
#include <math.h>

typedef __attribute__((ext_vector_type(8)))  int       v8i;
typedef __attribute__((ext_vector_type(8)))  float     v8f;
typedef __attribute__((ext_vector_type(16))) _Float16  v16h;
typedef __attribute__((ext_vector_type(8)))  _Float16  v8h;

#define B_   2
#define T_   2048
#define D_   2048
#define H_   32
#define KV_  8
#define HD_  64
#define TOK_ (B_ * T_)
#define BSTRIDE 80
#define NSTEP (D_ / 64)

__device__ inline v16h cat16(v8h lo, v8h hi) {
  return __builtin_shufflevector(lo, hi, 0,1,2,3,4,5,6,7,8,9,10,11,12,13,14,15);
}

__device__ inline void async_copy_b128(unsigned lds_off, const void* gaddr) {
  asm volatile("global_load_async_to_lds_b128 %0, %1, off"
               :: "v"(lds_off), "v"(gaddr) : "memory");
}
__device__ inline void wait_asynccnt0() {
  asm volatile("s_wait_asynccnt 0x0" ::: "memory");
}

__device__ inline v8i load_a_i8(const signed char* __restrict__ rowp, int hh) {
  v8i a;
#pragma unroll
  for (int j = 0; j < 4; ++j) {
    int2 t = *reinterpret_cast<const int2*>(rowp + (j >> 1) * 32 + hh * 16 + (j & 1) * 8);
    a[2 * j] = t.x; a[2 * j + 1] = t.y;
  }
  return a;
}

__device__ inline v8i load_b_lds(const signed char* rowp, int hh) {
  v8i b;
#pragma unroll
  for (int g = 0; g < 2; ++g) {
    int4 t = *reinterpret_cast<const int4*>(rowp + g * 32 + hh * 16);
    b[4 * g + 0] = t.x; b[4 * g + 1] = t.y; b[4 * g + 2] = t.z; b[4 * g + 3] = t.w;
  }
  return b;
}

__device__ inline v8i wmma_iu8(v8i a, v8i b, v8i c) {
  return __builtin_amdgcn_wmma_i32_16x16x64_iu8(true, a, true, b, c, false, false);
}

__device__ inline void gemm_step_schedule() {
  __builtin_amdgcn_sched_group_barrier(0x020, 5, 0);
  __builtin_amdgcn_sched_group_barrier(0x100, 4, 0);
#pragma unroll
  for (int k = 0; k < 6; ++k) {
    __builtin_amdgcn_sched_group_barrier(0x008, 1, 0);
    __builtin_amdgcn_sched_group_barrier(0x100, 2, 0);
  }
  __builtin_amdgcn_sched_group_barrier(0x008, 1, 0);
  __builtin_amdgcn_sched_group_barrier(0x008, 1, 0);
  __builtin_amdgcn_sched_barrier(0);
}

__global__ void absum_partial(const float* __restrict__ w, int n,
                              float* __restrict__ partial) {
  __shared__ float sm[256];
  float s = 0.f;
  for (int i = blockIdx.x * 256 + threadIdx.x; i < n; i += 256 * 256)
    s += fabsf(w[i]);
  sm[threadIdx.x] = s; __syncthreads();
  for (int off = 128; off > 0; off >>= 1) {
    if ((int)threadIdx.x < off) sm[threadIdx.x] += sm[threadIdx.x + off];
    __syncthreads();
  }
  if (threadIdx.x == 0) partial[blockIdx.x] = sm[0];
}

__global__ void absum_finalize(const float* __restrict__ partial,
                               float* __restrict__ wscale) {
  __shared__ float sm[256];
  const float counts[4] = {4194304.f, 1048576.f, 1048576.f, 4194304.f};
  for (int j = 0; j < 4; ++j) {
    sm[threadIdx.x] = partial[j * 256 + threadIdx.x];
    __syncthreads();
    for (int off = 128; off > 0; off >>= 1) {
      if ((int)threadIdx.x < off) sm[threadIdx.x] += sm[threadIdx.x + off];
      __syncthreads();
    }
    if (threadIdx.x == 0) wscale[j] = fmaxf(sm[0] / counts[j], 1e-5f);
    __syncthreads();
  }
}

__global__ void ternary_quant(const float* __restrict__ w, int n,
                              const float* __restrict__ wscale, int j,
                              signed char* __restrict__ out) {
  int i = blockIdx.x * 256 + threadIdx.x;
  if (i < n) {
    float q = rintf(w[i] / wscale[j]);
    q = fminf(1.f, fmaxf(-1.f, q));
    out[i] = (signed char)(int)q;
  }
}

__global__ void rmsnorm_quant(const float* __restrict__ x,
                              const float* __restrict__ g,
                              signed char* __restrict__ xq,
                              float* __restrict__ xs) {
  __shared__ float sm[256];
  const int tok = blockIdx.x;
  const float* xr = x + (size_t)tok * D_;
  float v[8];
  float ss = 0.f;
#pragma unroll
  for (int i = 0; i < 8; ++i) { v[i] = xr[threadIdx.x + i * 256]; ss += v[i] * v[i]; }
  sm[threadIdx.x] = ss; __syncthreads();
  for (int off = 128; off > 0; off >>= 1) {
    if ((int)threadIdx.x < off) sm[threadIdx.x] += sm[threadIdx.x + off];
    __syncthreads();
  }
  const float r = rsqrtf(sm[0] / (float)D_ + 1e-6f);
  __syncthreads();
  float mx = 0.f;
#pragma unroll
  for (int i = 0; i < 8; ++i) {
    v[i] = v[i] * r * g[threadIdx.x + i * 256];
    mx = fmaxf(mx, fabsf(v[i]));
  }
  sm[threadIdx.x] = mx; __syncthreads();
  for (int off = 128; off > 0; off >>= 1) {
    if ((int)threadIdx.x < off) sm[threadIdx.x] = fmaxf(sm[threadIdx.x], sm[threadIdx.x + off]);
    __syncthreads();
  }
  const float xscale = fmaxf(sm[0], 1e-5f);
  if (threadIdx.x == 0) xs[tok] = xscale;
  const float inv = 127.f / xscale;
  signed char* qr = xq + (size_t)tok * D_;
#pragma unroll
  for (int i = 0; i < 8; ++i) {
    float q = rintf(v[i] * inv);
    q = fminf(127.f, fmaxf(-128.f, q));
    qr[threadIdx.x + i * 256] = (signed char)(int)q;
  }
}

__global__ void __launch_bounds__(256) qkv_gemm_rope(
    const signed char* __restrict__ xq_q, const signed char* __restrict__ xq_k,
    const signed char* __restrict__ xq_v,
    const float* __restrict__ xs_q, const float* __restrict__ xs_k,
    const float* __restrict__ xs_v,
    const signed char* __restrict__ wq8, const signed char* __restrict__ wk8,
    const signed char* __restrict__ wv8,
    const float* __restrict__ wscale,
    const float* __restrict__ cos_p, const float* __restrict__ sin_p,
    _Float16* __restrict__ qbuf, _Float16* __restrict__ kbuf,
    _Float16* __restrict__ vbuf) {
  __shared__ __align__(16) signed char blds[2][128][BSTRIDE];
  const int tid = threadIdx.x;
  const int lane = tid & 31, wave = tid >> 5;
  const int n = lane & 15, hh = lane >> 4;
  const int gcol0 = blockIdx.y * 128;

  int which, o0;
  const signed char* xqs; const float* xss; const signed char* wsel;
  if (gcol0 < 2048)      { which = 0; xqs = xq_q; xss = xs_q; wsel = wq8; o0 = gcol0; }
  else if (gcol0 < 2560) { which = 1; xqs = xq_k; xss = xs_k; wsel = wk8; o0 = gcol0 - 2048; }
  else                   { which = 2; xqs = xq_v; xss = xs_v; wsel = wv8; o0 = gcol0 - 2560; }
  const float ws = wscale[which];
  const int row0 = blockIdx.x * 128 + wave * 16;

  const int cc = tid >> 1, cb = (tid & 1) * 32;
  const signed char* wrow = wsel + (size_t)(o0 + cc) * D_ + cb;
  auto stage_b = [&](int buf, int kk) {
    const unsigned lds = (unsigned)(uintptr_t)&blds[buf][cc][cb];
    async_copy_b128(lds,      wrow + kk);
    async_copy_b128(lds + 16, wrow + kk + 16);
  };

  v8i acc[8] = {};
  const signed char* ap = xqs + (size_t)(row0 + n) * D_;

  auto compute_step = [&](int i) {
    const int cur = i & 1;
    __builtin_prefetch(ap + i * 64 + 64, 0, 1);
    v8i a = load_a_i8(ap + i * 64, hh);
    v8i bf[8];
#pragma unroll
    for (int nt = 0; nt < 8; ++nt)
      bf[nt] = load_b_lds(&blds[cur][nt * 16 + n][0], hh);
#pragma unroll
    for (int nt = 0; nt < 8; ++nt)
      acc[nt] = wmma_iu8(a, bf[nt], acc[nt]);
    gemm_step_schedule();
  };

  stage_b(0, 0);
  wait_asynccnt0();
  __syncthreads();

  for (int i = 0; i < NSTEP - 1; ++i) {
    stage_b((i + 1) & 1, (i + 1) * 64);
    compute_step(i);
    wait_asynccnt0();
    __syncthreads();
  }
  compute_step(NSTEP - 1);

  float xsc[8]; int tokv[8];
#pragma unroll
  for (int v = 0; v < 8; ++v) {
    tokv[v] = row0 + v + 8 * hh;
    xsc[v] = xss[tokv[v]] * ws * (1.f / 127.f);
  }

  if (which < 2) {
#pragma unroll
    for (int pr = 0; pr < 8; ++pr) {
      if (pr & 2) continue;
      const int ntA = pr, ntB = pr ^ 2;
      const int lcA = o0 + ntA * 16 + n;
      const int lcB = lcA ^ 32;
      const int head = lcA >> 6;
      const int cA = lcA & 63, cB = lcB & 63;
#pragma unroll
      for (int v = 0; v < 8; ++v) {
        const float vA = (float)acc[ntA][v] * xsc[v];
        const float vB = (float)acc[ntB][v] * xsc[v];
        const int t = tokv[v] & (T_ - 1);
        const int b = tokv[v] >> 11;
        const float rA = vA * cos_p[t * HD_ + cA] - vB * sin_p[t * HD_ + cA];
        const float rB = vB * cos_p[t * HD_ + cB] + vA * sin_p[t * HD_ + cB];
        if (which == 0) {
          _Float16* dst = qbuf + ((size_t)(b * H_ + head) * T_ + t) * HD_;
          dst[cA] = (_Float16)rA;
          dst[cB] = (_Float16)rB;
        } else {
          _Float16* dst = kbuf + ((size_t)(b * KV_ + head) * T_ + t) * HD_;
          dst[cA] = (_Float16)rA;
          dst[cB] = (_Float16)rB;
        }
      }
    }
  } else {
#pragma unroll
    for (int nt = 0; nt < 8; ++nt) {
      const int lc = o0 + nt * 16 + n;
      const int kv = lc >> 6, c = lc & 63;
#pragma unroll
      for (int v = 0; v < 8; ++v) {
        const int t = tokv[v] & (T_ - 1);
        const int b = tokv[v] >> 11;
        vbuf[((size_t)(b * KV_ + kv) * HD_ + c) * T_ + t] =
            (_Float16)((float)acc[nt][v] * xsc[v]);
      }
    }
  }
}

__global__ void __launch_bounds__(128) attention_fa(
    const _Float16* __restrict__ qbuf, const _Float16* __restrict__ kbuf,
    const _Float16* __restrict__ vbuf, float* __restrict__ attn) {
  __shared__ _Float16 pbuf[4][16][32];
  const int lane = threadIdx.x & 31, wave = threadIdx.x >> 5;
  const int n = lane & 15, hh = lane >> 4;
  const int bh = blockIdx.y, b = bh >> 5, h = bh & 31, kvh = h >> 2;
  const int qbase = blockIdx.x * 64 + wave * 16;

  const _Float16* qrow = qbuf + ((size_t)(b * H_ + h) * T_ + (qbase + n)) * HD_;
  v16h qA0 = cat16(*reinterpret_cast<const v8h*>(qrow + hh * 8),
                   *reinterpret_cast<const v8h*>(qrow + 16 + hh * 8));
  v16h qA1 = cat16(*reinterpret_cast<const v8h*>(qrow + 32 + hh * 8),
                   *reinterpret_cast<const v8h*>(qrow + 48 + hh * 8));
#pragma unroll
  for (int i = 0; i < 16; ++i) {
    qA0[i] = (_Float16)((float)qA0[i] * 0.125f);
    qA1[i] = (_Float16)((float)qA1[i] * 0.125f);
  }

  v8f o_acc[4] = {};
  float m_i[8], l_i[8];
#pragma unroll
  for (int v = 0; v < 8; ++v) { m_i[v] = -3.0e38f; l_i[v] = 0.f; }

  const _Float16* kbase_p = kbuf + (size_t)(b * KV_ + kvh) * T_ * HD_;
  const _Float16* vbase_p = vbuf + (size_t)(b * KV_ + kvh) * HD_ * T_;
  const int nkv = qbase + 16;

  for (int kv0 = 0; kv0 < nkv; kv0 += 32) {
    v8f s[2];
#pragma unroll
    for (int sub = 0; sub < 2; ++sub) {
      const _Float16* krow = kbase_p + (size_t)(kv0 + sub * 16 + n) * HD_;
      v16h kB0 = cat16(*reinterpret_cast<const v8h*>(krow + hh * 8),
                       *reinterpret_cast<const v8h*>(krow + 16 + hh * 8));
      v16h kB1 = cat16(*reinterpret_cast<const v8h*>(krow + 32 + hh * 8),
                       *reinterpret_cast<const v8h*>(krow + 48 + hh * 8));
      v8f z = {};
      v8f t0 = __builtin_amdgcn_wmma_f32_16x16x32_f16(false, qA0, false, kB0,
                                                      (short)0, z, false, false);
      s[sub] = __builtin_amdgcn_wmma_f32_16x16x32_f16(false, qA1, false, kB1,
                                                      (short)0, t0, false, false);
    }

    float p0[8], p1[8], alpha[8];
#pragma unroll
    for (int v = 0; v < 8; ++v) {
      const int row = qbase + v + 8 * hh;
      float s0 = (kv0 + n      <= row) ? s[0][v] : -3.0e38f;
      float s1 = (kv0 + 16 + n <= row) ? s[1][v] : -3.0e38f;
      float mx = fmaxf(s0, s1);
#pragma unroll
      for (int msk = 8; msk >= 1; msk >>= 1) mx = fmaxf(mx, __shfl_xor(mx, msk, 32));
      const float mnew = fmaxf(m_i[v], mx);
      alpha[v] = __expf(m_i[v] - mnew);
      p0[v] = __expf(s0 - mnew);
      p1[v] = __expf(s1 - mnew);
      float sum = p0[v] + p1[v];
#pragma unroll
      for (int msk = 8; msk >= 1; msk >>= 1) sum += __shfl_xor(sum, msk, 32);
      l_i[v] = l_i[v] * alpha[v] + sum;
      m_i[v] = mnew;
    }
#pragma unroll
    for (int nt = 0; nt < 4; ++nt)
#pragma unroll
      for (int v = 0; v < 8; ++v) o_acc[nt][v] *= alpha[v];

#pragma unroll
    for (int v = 0; v < 8; ++v) {
      const int m = v + 8 * hh;
      pbuf[wave][m][n]      = (_Float16)p0[v];
      pbuf[wave][m][16 + n] = (_Float16)p1[v];
    }
    asm volatile("s_wait_dscnt 0" ::: "memory");
    const _Float16* prow = &pbuf[wave][n][0];
    v16h pA = cat16(*reinterpret_cast<const v8h*>(prow + hh * 8),
                    *reinterpret_cast<const v8h*>(prow + 16 + hh * 8));

#pragma unroll
    for (int nt = 0; nt < 4; ++nt) {
      const _Float16* vcol = vbase_p + (size_t)(nt * 16 + n) * T_ + kv0;
      v16h vB = cat16(*reinterpret_cast<const v8h*>(vcol + hh * 8),
                      *reinterpret_cast<const v8h*>(vcol + 16 + hh * 8));
      o_acc[nt] = __builtin_amdgcn_wmma_f32_16x16x32_f16(false, pA, false, vB,
                                                         (short)0, o_acc[nt],
                                                         false, false);
    }
  }

#pragma unroll
  for (int nt = 0; nt < 4; ++nt) {
    const int c = h * HD_ + nt * 16 + n;
#pragma unroll
    for (int v = 0; v < 8; ++v) {
      const int row = qbase + v + 8 * hh;
      attn[((size_t)(b * T_ + row)) * D_ + c] = o_acc[nt][v] / l_i[v];
    }
  }
}

__global__ void __launch_bounds__(256) out_gemm(
    const signed char* __restrict__ xq8, const float* __restrict__ xs,
    const signed char* __restrict__ w8, const float* __restrict__ wscale,
    float* __restrict__ out) {
  __shared__ __align__(16) signed char blds[2][128][BSTRIDE];
  const int tid = threadIdx.x;
  const int lane = tid & 31, wave = tid >> 5;
  const int n = lane & 15, hh = lane >> 4;
  const int o0 = blockIdx.y * 128;
  const int row0 = blockIdx.x * 128 + wave * 16;
  const float ws = wscale[3];

  const int cc = tid >> 1, cb = (tid & 1) * 32;
  const signed char* wrow = w8 + (size_t)(o0 + cc) * D_ + cb;
  auto stage_b = [&](int buf, int kk) {
    const unsigned lds = (unsigned)(uintptr_t)&blds[buf][cc][cb];
    async_copy_b128(lds,      wrow + kk);
    async_copy_b128(lds + 16, wrow + kk + 16);
  };

  v8i acc[8] = {};
  const signed char* ap = xq8 + (size_t)(row0 + n) * D_;

  auto compute_step = [&](int i) {
    const int cur = i & 1;
    __builtin_prefetch(ap + i * 64 + 64, 0, 1);
    v8i a = load_a_i8(ap + i * 64, hh);
    v8i bf[8];
#pragma unroll
    for (int nt = 0; nt < 8; ++nt)
      bf[nt] = load_b_lds(&blds[cur][nt * 16 + n][0], hh);
#pragma unroll
    for (int nt = 0; nt < 8; ++nt)
      acc[nt] = wmma_iu8(a, bf[nt], acc[nt]);
    gemm_step_schedule();
  };

  stage_b(0, 0);
  wait_asynccnt0();
  __syncthreads();

  for (int i = 0; i < NSTEP - 1; ++i) {
    stage_b((i + 1) & 1, (i + 1) * 64);
    compute_step(i);
    wait_asynccnt0();
    __syncthreads();
  }
  compute_step(NSTEP - 1);

#pragma unroll
  for (int v = 0; v < 8; ++v) {
    const int tok = row0 + v + 8 * hh;
    const float sc = xs[tok] * ws * (1.f / 127.f);
#pragma unroll
    for (int nt = 0; nt < 8; ++nt)
      out[(size_t)tok * D_ + o0 + nt * 16 + n] = (float)acc[nt][v] * sc;
  }
}

extern "C" void kernel_launch(void* const* d_in, const int* in_sizes, int n_in,
                              void* d_out, int out_size, void* d_ws, size_t ws_size,
                              hipStream_t stream) {
  const float* x    = (const float*)d_in[0];
  const float* cosp = (const float*)d_in[1];
  const float* sinp = (const float*)d_in[2];
  const float* wq   = (const float*)d_in[3];
  const float* wk   = (const float*)d_in[4];
  const float* wv   = (const float*)d_in[5];
  const float* wo   = (const float*)d_in[6];
  const float* gq   = (const float*)d_in[7];
  const float* gk   = (const float*)d_in[8];
  const float* gv   = (const float*)d_in[9];
  const float* go   = (const float*)d_in[10];
  float* out = (float*)d_out;

  char* ws = (char*)d_ws;
  size_t off = 0;
  auto alloc = [&](size_t bytes) -> char* {
    off = (off + 255) & ~(size_t)255;
    char* p = ws + off;
    off += bytes;
    return p;
  };

  float* partials = (float*)alloc(4 * 256 * sizeof(float));
  float* wscale   = (float*)alloc(4 * sizeof(float));
  signed char* wq8 = (signed char*)alloc((size_t)2048 * 2048);
  signed char* wk8 = (signed char*)alloc((size_t)512 * 2048);
  signed char* wv8 = (signed char*)alloc((size_t)512 * 2048);
  signed char* wo8 = (signed char*)alloc((size_t)2048 * 2048);
  signed char* xqQ = (signed char*)alloc((size_t)TOK_ * D_);
  signed char* xqK = (signed char*)alloc((size_t)TOK_ * D_);
  signed char* xqV = (signed char*)alloc((size_t)TOK_ * D_);
  signed char* xqO = (signed char*)alloc((size_t)TOK_ * D_);
  float* xsQ = (float*)alloc(TOK_ * sizeof(float));
  float* xsK = (float*)alloc(TOK_ * sizeof(float));
  float* xsV = (float*)alloc(TOK_ * sizeof(float));
  float* xsO = (float*)alloc(TOK_ * sizeof(float));
  _Float16* qbuf = (_Float16*)alloc((size_t)TOK_ * D_ * 2);
  _Float16* kbuf = (_Float16*)alloc((size_t)B_ * KV_ * T_ * HD_ * 2);
  _Float16* vbuf = (_Float16*)alloc((size_t)B_ * KV_ * T_ * HD_ * 2);
  float* attn = (float*)alloc((size_t)TOK_ * D_ * sizeof(float));

  absum_partial<<<256, 256, 0, stream>>>(wq, 2048 * 2048, partials + 0 * 256);
  absum_partial<<<256, 256, 0, stream>>>(wk, 512 * 2048,  partials + 1 * 256);
  absum_partial<<<256, 256, 0, stream>>>(wv, 512 * 2048,  partials + 2 * 256);
  absum_partial<<<256, 256, 0, stream>>>(wo, 2048 * 2048, partials + 3 * 256);
  absum_finalize<<<1, 256, 0, stream>>>(partials, wscale);

  ternary_quant<<<(4194304 + 255) / 256, 256, 0, stream>>>(wq, 4194304, wscale, 0, wq8);
  ternary_quant<<<(1048576 + 255) / 256, 256, 0, stream>>>(wk, 1048576, wscale, 1, wk8);
  ternary_quant<<<(1048576 + 255) / 256, 256, 0, stream>>>(wv, 1048576, wscale, 2, wv8);
  ternary_quant<<<(4194304 + 255) / 256, 256, 0, stream>>>(wo, 4194304, wscale, 3, wo8);

  rmsnorm_quant<<<TOK_, 256, 0, stream>>>(x, gq, xqQ, xsQ);
  rmsnorm_quant<<<TOK_, 256, 0, stream>>>(x, gk, xqK, xsK);
  rmsnorm_quant<<<TOK_, 256, 0, stream>>>(x, gv, xqV, xsV);

  qkv_gemm_rope<<<dim3(TOK_ / 128, 3072 / 128), 256, 0, stream>>>(
      xqQ, xqK, xqV, xsQ, xsK, xsV, wq8, wk8, wv8, wscale, cosp, sinp,
      qbuf, kbuf, vbuf);

  attention_fa<<<dim3(T_ / 64, B_ * H_), 128, 0, stream>>>(qbuf, kbuf, vbuf, attn);

  rmsnorm_quant<<<TOK_, 256, 0, stream>>>(attn, go, xqO, xsO);
  out_gemm<<<dim3(TOK_ / 128, 2048 / 128), 256, 0, stream>>>(xqO, xsO, wo8, wscale, out);
}
